// TemporalLSTM_88725434401477
// MI455X (gfx1250) — compile-verified, matrix operand order checked by reading, not run on hardware
//
#include <hip/hip_runtime.h>

#define Bc 512
#define Sc 128
#define Dc 256
#define HSc 512
#define Oc 64
#define Gc (4*HSc)

typedef __attribute__((ext_vector_type(16))) __bf16 v16bf;
typedef __attribute__((ext_vector_type(8)))  __bf16 v8bf;
typedef __attribute__((ext_vector_type(8)))  float  v8f;

__device__ __forceinline__ unsigned short f2bf(float f){
  unsigned int u = __float_as_uint(f);
  u += 0x7FFFu + ((u >> 16) & 1u);          // round-to-nearest-even
  return (unsigned short)(u >> 16);
}

// Load one 16-bf16 A/B fragment for v_wmma_f32_16x16x32_bf16.
// Caller passes p already offset to row(lane&15)*ld + kbase + ((lane>>4)*8).
// Lane l<16 holds K[0..7],K[16..23]; lane l>=16 holds K[8..15],K[24..31].
__device__ __forceinline__ v16bf ld_frag(const unsigned short* p){
  v8bf lo = *(const v8bf*)(p);
  v8bf hi = *(const v8bf*)(p + 16);
  v16bf r;
#pragma unroll
  for (int i = 0; i < 8; ++i){ r[i] = lo[i]; r[i+8] = hi[i]; }
  return r;
}

__device__ __forceinline__ v8f wmma_bf16(v16bf a, v16bf b, v8f c){
  return __builtin_amdgcn_wmma_f32_16x16x32_bf16(false, a, false, b, (short)0, c, false, false);
}

// gfx1250 async copy: 16 bytes global -> LDS, tracked by ASYNCcnt.
__device__ __forceinline__ void async_cp16(unsigned lds_off, const unsigned short* gp){
  asm volatile("global_load_async_to_lds_b128 %0, %1, off"
               :: "v"(lds_off), "v"((unsigned long long)(uintptr_t)gp) : "memory");
}
__device__ __forceinline__ void async_wait0(){
  asm volatile("s_wait_asynccnt 0x0" ::: "memory");
}

// ---------------- one-time conversion kernels ----------------

__global__ void k_conv_H(const float* __restrict__ H,
                         unsigned short* __restrict__ Hbf,
                         unsigned short* __restrict__ HbfT){
  for (size_t idx = (size_t)blockIdx.x*blockDim.x + threadIdx.x;
       idx < (size_t)Bc*Sc*Dc; idx += (size_t)gridDim.x*blockDim.x){
    int d = (int)(idx % Dc);
    size_t r = idx / Dc;
    int s = (int)(r % Sc);
    int b = (int)(r / Sc);
    unsigned short v = f2bf(H[idx]);
    Hbf[idx] = v;                                   // [b][s][d]
    HbfT[((size_t)b*Dc + d)*Sc + s] = v;            // [b][d][s]
  }
}

__global__ void k_transpose(const float* __restrict__ src,
                            unsigned short* __restrict__ dst, int R, int C){
  for (size_t idx = (size_t)blockIdx.x*blockDim.x + threadIdx.x;
       idx < (size_t)R*C; idx += (size_t)gridDim.x*blockDim.x){
    int c = (int)(idx % C);
    int r = (int)(idx / C);
    dst[(size_t)c*R + r] = f2bf(src[idx]);          // [C][R] (column-major of src)
  }
}

__global__ void k_convert(const float* __restrict__ src,
                          unsigned short* __restrict__ dst, int n){
  for (int idx = blockIdx.x*blockDim.x + threadIdx.x; idx < n;
       idx += gridDim.x*blockDim.x)
    dst[idx] = f2bf(src[idx]);
}

// UF^T[g][hs] = bf16( U[hs][g] + sum_o fc_w[o][hs]*Wy[o][g] )   (Wy path folded into U)
__global__ void k_prep_uf(const float* __restrict__ U, const float* __restrict__ fcw,
                          const float* __restrict__ Wy, unsigned short* __restrict__ UFT){
  int idx = blockIdx.x*blockDim.x + threadIdx.x;     // covers Gc*HSc
  int hs = idx & (HSc-1), g = idx >> 9;
  float acc = U[(size_t)hs*Gc + g];
  for (int o = 0; o < Oc; ++o)
    acc += fcw[(size_t)o*HSc + hs] * Wy[(size_t)o*Gc + g];
  UFT[(size_t)g*HSc + hs] = f2bf(acc);
}

// bias2[g] = bias[g] + sum_o fc_b[o]*Wy[o][g]
__global__ void k_prep_bias2(const float* __restrict__ bias, const float* __restrict__ fcb,
                             const float* __restrict__ Wy, float* __restrict__ bias2){
  int g = blockIdx.x*blockDim.x + threadIdx.x;
  if (g >= Gc) return;
  float acc = bias[g];
  for (int o = 0; o < Oc; ++o) acc += fcb[o] * Wy[(size_t)o*Gc + g];
  bias2[g] = acc;
}

// g0y[b][g] = y_t_1[b]@Wy  (only needed for step 0)
__global__ void k_prep_g0y(const float* __restrict__ y0, const float* __restrict__ Wy,
                           float* __restrict__ g0y){
  int idx = blockIdx.x*blockDim.x + threadIdx.x;     // covers Bc*Gc
  int g = idx & (Gc-1), b = idx >> 11;
  float acc = 0.f;
  for (int o = 0; o < Oc; ++o) acc += y0[(size_t)b*Oc + o] * Wy[(size_t)o*Gc + g];
  g0y[(size_t)b*Gc + g] = acc;
}

__global__ void k_init(unsigned short* __restrict__ hA,
                       unsigned short* __restrict__ hB,
                       float* __restrict__ cbuf){
  int idx = blockIdx.x*blockDim.x + threadIdx.x;     // grid covers Bc*HSc
  hA[idx] = 0; hB[idx] = 0; cbuf[idx] = 0.f;
}

// ---------------- attention: e = tanh(H_t@Wa'+ba)@Va' ; softmax ----------------
// grid = S * (B/16) blocks, 256 threads (8 waves). wave w owns n-tile w.

__global__ __launch_bounds__(256) void k_attention(
    const unsigned short* __restrict__ Hbf, const unsigned short* __restrict__ WaT,
    const unsigned short* __restrict__ VaT, const float* __restrict__ ba,
    float* __restrict__ betas_out, unsigned short* __restrict__ betasbf){
  __shared__ __align__(16) unsigned short tanhT[16*Sc];
  __shared__ float eT[16*Sc];
  __shared__ float red[32];
  int t    = blockIdx.x >> 5;
  int bb   = blockIdx.x & 31;
  int wave = threadIdx.x >> 5, lane = threadIdx.x & 31;
  int mr = lane & 15, koff = (lane >> 4) << 3, mb = koff;
  int ncol = wave*16 + mr;

  v8f acc = {};
  const unsigned short* ap = Hbf + ((size_t)(bb*16 + mr)*Sc + t)*Dc + koff;
  const unsigned short* bp = WaT + (size_t)ncol*Dc + koff;
  for (int k = 0; k < Dc; k += 32)
    acc = wmma_bf16(ld_frag(ap + k), ld_frag(bp + k), acc);
  float bav = ba[ncol];
#pragma unroll
  for (int v = 0; v < 8; ++v)
    tanhT[(mb + v)*Sc + ncol] = f2bf(tanhf(acc[v] + bav));
  __syncthreads();

  v8f acc2 = {};
  const unsigned short* ap2 = tanhT + mr*Sc + koff;
  const unsigned short* bp2 = VaT + (size_t)ncol*Sc + koff;
  for (int k = 0; k < Sc; k += 32)
    acc2 = wmma_bf16(ld_frag(ap2 + k), ld_frag(bp2 + k), acc2);
#pragma unroll
  for (int v = 0; v < 8; ++v) eT[(mb + v)*Sc + ncol] = acc2[v];
  __syncthreads();

  if (threadIdx.x < 16){
    int r = threadIdx.x;
    float m = -1e30f;
    for (int j = 0; j < Sc; ++j) m = fmaxf(m, eT[r*Sc + j]);
    float s = 0.f;
    for (int j = 0; j < Sc; ++j) s += __expf(eT[r*Sc + j] - m);
    red[r] = m; red[16 + r] = 1.f / s;
  }
  __syncthreads();
  for (int idx = threadIdx.x; idx < 16*Sc; idx += 256){
    int r = idx >> 7, j = idx & (Sc - 1);
    float v = __expf(eT[idx] - red[r]) * red[16 + r];
    size_t o = ((size_t)t*Bc + (bb*16 + r))*Sc + j;
    betas_out[o] = v;
    betasbf[o]   = f2bf(v);
  }
}

// ---------------- ctx[t][b][:] = Betas[:,b,:] @ H[b] (per-b GEMM) ----------------

__global__ __launch_bounds__(256) void k_context(
    const unsigned short* __restrict__ betasbf,
    const unsigned short* __restrict__ HbfT,
    unsigned short* __restrict__ ctxbf){
  int b  = blockIdx.x >> 3;
  int t0 = (blockIdx.x & 7) * 16;
  int wave = threadIdx.x >> 5, lane = threadIdx.x & 31;
  int mr = lane & 15, koff = (lane >> 4) << 3, mb = koff;
  const unsigned short* ap = betasbf + ((size_t)(t0 + mr)*Bc + b)*Sc + koff;
  for (int dt = wave; dt < Dc/16; dt += 8){
    int dcol = dt*16 + mr;
    const unsigned short* bp = HbfT + ((size_t)b*Dc + dcol)*Sc + koff;
    v8f acc = {};
    for (int k = 0; k < Sc; k += 32)
      acc = wmma_bf16(ld_frag(ap + k), ld_frag(bp + k), acc);
#pragma unroll
    for (int v = 0; v < 8; ++v)
      ctxbf[((size_t)(t0 + mb + v)*Bc + b)*Dc + dcol] = f2bf(acc[v]);
  }
}

// ---------------- fused per-step kernel ----------------
// gates = ctx_t@W + h_prev@UF + bias(+extra at t=0), then LSTM cell, all in-block.
// grid = (B/32) * (HS/16) = 512 blocks, 8 waves.
// Block owns 2 m-tiles (32 b-rows) and one 16-wide j-slice; wave (mi,q) computes
// gate tile q (i/f/g/o) for m-tile mi. A-panels staged to LDS via async copies.

__global__ __launch_bounds__(256) void k_step(
    const unsigned short* __restrict__ ctxbf_t,   // [B][D]
    const unsigned short* __restrict__ hprev,     // [B][HS]
    const unsigned short* __restrict__ WT,        // [G][D]
    const unsigned short* __restrict__ UFT,       // [G][HS]
    const float* __restrict__ biasv,              // [G]
    const float* __restrict__ extra,              // [B][G] or nullptr
    float* __restrict__ cbuf,
    unsigned short* __restrict__ hnext,           // [B][HS]
    float* __restrict__ hidden_out, int t){
  __shared__ __align__(16) unsigned short Ah[32*HSc];    // 32 KB: h rows
  __shared__ __align__(16) unsigned short Actx[32*Dc];   // 16 KB: ctx rows
  __shared__ float gtile[2*4*16*16];                     // 8 KB: i/f/g/o tiles
  int mp = blockIdx.x & 15;           // m-pair (32 rows of b)
  int j0 = (blockIdx.x >> 4) * 16;    // j-slice
  int wave = threadIdx.x >> 5, lane = threadIdx.x & 31;
  int mr = lane & 15, koff = (lane >> 4) << 3, mb = koff;

  // ---- async stage A panels (global -> LDS) ----
  const unsigned short* hsrc = hprev   + (size_t)mp*32*HSc;
  const unsigned short* csrc = ctxbf_t + (size_t)mp*32*Dc;
  for (int i = threadIdx.x; i < 32*HSc/8; i += 256){      // 2048 chunks of 16B
    int r = i >> 6, cc = (i & 63) << 3;
    async_cp16((unsigned)(uintptr_t)(Ah + r*HSc + cc), hsrc + (size_t)r*HSc + cc);
  }
  for (int i = threadIdx.x; i < 32*Dc/8; i += 256){       // 1024 chunks of 16B
    int r = i >> 5, cc = (i & 31) << 3;
    async_cp16((unsigned)(uintptr_t)(Actx + r*Dc + cc), csrc + (size_t)r*Dc + cc);
  }
  async_wait0();
  __syncthreads();

  // ---- WMMA: one gate tile per wave ----
  int mi = wave >> 2, q = wave & 3;
  int m0 = (mp*2 + mi)*16;
  int ncol = q*HSc + j0 + mr;
  v8f acc = {};
  { const unsigned short* a = Actx + (mi*16 + mr)*Dc + koff;
    const unsigned short* b = WT + (size_t)ncol*Dc + koff;
    for (int k = 0; k < Dc; k += 32)
      acc = wmma_bf16(ld_frag(a + k), ld_frag(b + k), acc); }
  { const unsigned short* a = Ah + (mi*16 + mr)*HSc + koff;
    const unsigned short* b = UFT + (size_t)ncol*HSc + koff;
    for (int k = 0; k < HSc; k += 32){
      __builtin_prefetch(b + k + 256, 0, 0);              // UF stays L2-hot
      acc = wmma_bf16(ld_frag(a + k), ld_frag(b + k), acc);
    } }
  float bv = biasv[ncol];
  if (extra){                                             // t==0: + y_t_1@Wy
#pragma unroll
    for (int v = 0; v < 8; ++v)
      acc[v] += extra[(size_t)(m0 + mb + v)*Gc + ncol];
  }
#pragma unroll
  for (int v = 0; v < 8; ++v)
    gtile[((mi*4 + q)*16 + (mb + v))*16 + mr] = acc[v] + bv;
  __syncthreads();

  // ---- LSTM cell elementwise (512 elements, 2 per thread) ----
  for (int id = threadIdx.x; id < 512; id += 256){
    int emi = id >> 8, r = (id >> 4) & 15, jc = id & 15;
    const float* g = &gtile[(emi*4)*16*16 + r*16 + jc];
    float it = 1.f/(1.f + __expf(-g[0*256]));
    float ft = 1.f/(1.f + __expf(-g[1*256]));
    float gt = tanhf(g[2*256]);
    float ot = 1.f/(1.f + __expf(-g[3*256]));
    int b = mp*32 + emi*16 + r;
    int j = j0 + jc;
    size_t cidx = (size_t)b*HSc + j;
    float c = ft*cbuf[cidx] + it*gt;
    float h = ot*tanhf(c);
    cbuf[cidx]  = c;
    hnext[cidx] = f2bf(h);
    hidden_out[((size_t)b*Sc + t)*HSc + j] = h;
  }
}

// ---------------- final y = h @ fc_w^T + fc_b ----------------

__global__ __launch_bounds__(256) void k_fc(
    const unsigned short* __restrict__ hbf, const unsigned short* __restrict__ fcw,
    const float* __restrict__ fcb, float* __restrict__ y_out){
  int wave = threadIdx.x >> 5, lane = threadIdx.x & 31;
  int tile = blockIdx.x*8 + wave;
  int m0 = (tile >> 2)*16, n0 = (tile & 3)*16;
  int mr = lane & 15, koff = (lane >> 4) << 3, mb = koff;
  int ncol = n0 + mr;
  v8f acc = {};
  const unsigned short* ap = hbf + (size_t)(m0 + mr)*HSc + koff;
  const unsigned short* bp = fcw + (size_t)ncol*HSc + koff;
  for (int k = 0; k < HSc; k += 32)
    acc = wmma_bf16(ld_frag(ap + k), ld_frag(bp + k), acc);
  float bv = fcb[ncol];
#pragma unroll
  for (int v = 0; v < 8; ++v)
    y_out[(size_t)(m0 + mb + v)*Oc + ncol] = acc[v] + bv;
}

extern "C" void kernel_launch(void* const* d_in, const int* in_sizes, int n_in,
                              void* d_out, int out_size, void* d_ws, size_t ws_size,
                              hipStream_t stream) {
  (void)in_sizes; (void)n_in; (void)out_size; (void)ws_size;
  const float* H    = (const float*)d_in[0];
  const float* y0   = (const float*)d_in[1];
  const float* Wa   = (const float*)d_in[2];
  /* d_in[3] = Ua: multiplied by zeros in the reference, unused */
  const float* ba   = (const float*)d_in[4];
  const float* Va   = (const float*)d_in[5];
  const float* W    = (const float*)d_in[6];
  const float* U    = (const float*)d_in[7];
  const float* bias = (const float*)d_in[8];
  const float* Wy   = (const float*)d_in[9];
  const float* fcw  = (const float*)d_in[10];
  const float* fcb  = (const float*)d_in[11];

  float* out        = (float*)d_out;
  float* y_out      = out;                                   // [B,O]
  float* hidden_out = out + (size_t)Bc*Oc;                   // [B,S,HS]
  float* betas_out  = hidden_out + (size_t)Bc*Sc*HSc;        // [S,B,S]

  char* ws = (char*)d_ws;
  size_t off = 0;
  auto alloc = [&](size_t bytes)->char*{
    char* p = ws + off; off += (bytes + 255) & ~(size_t)255; return p; };
  unsigned short* Hbf     = (unsigned short*)alloc((size_t)Bc*Sc*Dc*2);
  unsigned short* HbfT    = (unsigned short*)alloc((size_t)Bc*Sc*Dc*2);
  unsigned short* WaT     = (unsigned short*)alloc((size_t)Sc*Dc*2);
  unsigned short* VaT     = (unsigned short*)alloc((size_t)Sc*Sc*2);
  unsigned short* WT      = (unsigned short*)alloc((size_t)Gc*Dc*2);
  unsigned short* UFT     = (unsigned short*)alloc((size_t)Gc*HSc*2);
  unsigned short* FCW     = (unsigned short*)alloc((size_t)Oc*HSc*2);
  unsigned short* betasbf = (unsigned short*)alloc((size_t)Sc*Bc*Sc*2);
  unsigned short* ctxbf   = (unsigned short*)alloc((size_t)Sc*Bc*Dc*2);
  unsigned short* hA      = (unsigned short*)alloc((size_t)Bc*HSc*2);
  unsigned short* hB      = (unsigned short*)alloc((size_t)Bc*HSc*2);
  float*          cbuf    = (float*)alloc((size_t)Bc*HSc*4);
  float*          bias2   = (float*)alloc((size_t)Gc*4);
  float*          g0y     = (float*)alloc((size_t)Bc*Gc*4);

  // one-time: precision conversion, layouts, Wy-path folding
  k_conv_H    <<<2048, 256, 0, stream>>>(H, Hbf, HbfT);
  k_transpose <<< 128, 256, 0, stream>>>(Wa, WaT, Dc, Sc);
  k_transpose <<<  64, 256, 0, stream>>>(Va, VaT, Sc, Sc);
  k_transpose <<<2048, 256, 0, stream>>>(W,  WT,  Dc, Gc);
  k_convert   <<< 128, 256, 0, stream>>>(fcw, FCW, Oc*HSc);
  k_prep_uf   <<<Gc*HSc/256, 256, 0, stream>>>(U, fcw, Wy, UFT);
  k_prep_bias2<<<Gc/256, 256, 0, stream>>>(bias, fcb, Wy, bias2);
  k_prep_g0y  <<<Bc*Gc/256, 256, 0, stream>>>(y0, Wy, g0y);
  k_init      <<<Bc*HSc/256, 256, 0, stream>>>(hA, hB, cbuf);

  // t-parallel attention + context (~11 GFLOP vs ~206 for the recurrence)
  k_attention<<<Sc*(Bc/16), 256, 0, stream>>>(Hbf, WaT, VaT, ba, betas_out, betasbf);
  k_context  <<<Bc*(Sc/16), 256, 0, stream>>>(betasbf, HbfT, ctxbf);

  // sequential recurrence: ONE kernel per step (double-buffered h)
  unsigned short* hb[2] = { hA, hB };
  for (int t = 0; t < Sc; ++t){
    k_step<<<512, 256, 0, stream>>>(ctxbf + (size_t)t*Bc*Dc, hb[t & 1],
                                    WT, UFT,
                                    t == 0 ? bias : bias2,
                                    t == 0 ? g0y : (const float*)nullptr,
                                    cbuf, hb[(t + 1) & 1], hidden_out, t);
  }
  k_fc<<<16, 256, 0, stream>>>(hb[0], FCW, fcb, y_out);   // h after t=127 is in hA
}
